// GMAMDA_86517821210791
// MI455X (gfx1250) — matrix operand order repaired and checked by reading, not compiled, not run on hardware
//
#include <hip/hip_runtime.h>

typedef __bf16 bf16;
typedef __attribute__((ext_vector_type(8)))  __bf16 v8bf;
typedef __attribute__((ext_vector_type(16))) __bf16 v16bf;
typedef __attribute__((ext_vector_type(8)))  float  v8f;

#define NN    2249
#define NDRC  2033
#define NEC   300000
#define NBC   16384

__device__ __forceinline__ bf16 f2bf(float f) {
  unsigned u = __float_as_uint(f);
  unsigned r = (u + 0x7FFFu + ((u >> 16) & 1u)) >> 16;
  unsigned short s = (unsigned short)r;
  return __builtin_bit_cast(bf16, s);
}

__device__ __forceinline__ void atomAddF(float* p, float v) {
#if defined(__HIP_PLATFORM_AMD__)
  unsafeAtomicAdd(p, v);   // native global_atomic_add_f32
#else
  atomicAdd(p, v);
#endif
}

__device__ __forceinline__ v16bf loadAfrag(const bf16* p) {
  v8bf lo = *(const v8bf*)(p);        // K = hi*16 + 0..7
  v8bf hi = *(const v8bf*)(p + 8);    // K = hi*16 + 8..15
  return __builtin_shufflevector(lo, hi,
      0,1,2,3,4,5,6,7,8,9,10,11,12,13,14,15);
}

// ---------------------------------------------------------------------------
// bf16 WMMA GEMM, register-tiled: one wave computes a 32x64 C block
// (2 M-tiles x 4 N-tiles). Bfrag layout: [nt][kt][n:16][k:32].
// A must be padded to 32-row / 32-col multiples; stores are bounds-guarded.
// ---------------------------------------------------------------------------
__global__ __launch_bounds__(128) void gm_wmma_gemm(
    const bf16* __restrict__ A, int lda,
    const bf16* __restrict__ Bfrag,
    float* __restrict__ Cf, bf16* __restrict__ Cb,
    int ldc, int M, int Nbound,
    const float* __restrict__ bias,
    int Mgroups, int Ngroups, int Ktiles, int flags /*bit0: relu*/) {
  int wid = (int)((blockIdx.x * blockDim.x + threadIdx.x) >> 5);
  if (wid >= Mgroups * Ngroups) return;        // uniform per wave
  int mg = wid / Ngroups, ng = wid % Ngroups;
  int lane = threadIdx.x & 31;
  int n  = lane & 15;                          // col-in-tile; also A row-in-tile
  int hi = lane >> 4;

  const bf16* arow0 = A + (size_t)(mg * 32 + n) * lda + hi * 16;
  const bf16* arow1 = arow0 + (size_t)16 * lda;
  const size_t bStride = (size_t)Ktiles * 512;
  const bf16* bbase = Bfrag + (size_t)(ng * 4) * bStride + n * 32 + hi * 16;

  v8f zero = {};
  v8f c00 = zero, c01 = zero, c02 = zero, c03 = zero;
  v8f c10 = zero, c11 = zero, c12 = zero, c13 = zero;

  for (int kt = 0; kt < Ktiles; ++kt) {
    v16bf a0 = loadAfrag(arow0);
    v16bf a1 = loadAfrag(arow1);
    const bf16* bp = bbase + (size_t)kt * 512;
    v16bf b0 = *(const v16bf*)(bp);
    v16bf b1 = *(const v16bf*)(bp + bStride);
    v16bf b2 = *(const v16bf*)(bp + 2 * bStride);
    v16bf b3 = *(const v16bf*)(bp + 3 * bStride);
    c00 = __builtin_amdgcn_wmma_f32_16x16x32_bf16(false, a0, false, b0, (short)0, c00, false, false);
    c10 = __builtin_amdgcn_wmma_f32_16x16x32_bf16(false, a1, false, b0, (short)0, c10, false, false);
    c01 = __builtin_amdgcn_wmma_f32_16x16x32_bf16(false, a0, false, b1, (short)0, c01, false, false);
    c11 = __builtin_amdgcn_wmma_f32_16x16x32_bf16(false, a1, false, b1, (short)0, c11, false, false);
    c02 = __builtin_amdgcn_wmma_f32_16x16x32_bf16(false, a0, false, b2, (short)0, c02, false, false);
    c12 = __builtin_amdgcn_wmma_f32_16x16x32_bf16(false, a1, false, b2, (short)0, c12, false, false);
    c03 = __builtin_amdgcn_wmma_f32_16x16x32_bf16(false, a0, false, b3, (short)0, c03, false, false);
    c13 = __builtin_amdgcn_wmma_f32_16x16x32_bf16(false, a1, false, b3, (short)0, c13, false, false);
    arow0 += 32;
    arow1 += 32;
  }

  v8f acc[2][4] = {{c00, c01, c02, c03}, {c10, c11, c12, c13}};
#pragma unroll
  for (int tn = 0; tn < 4; ++tn) {
    int col = ng * 64 + tn * 16 + n;
    if (col < Nbound) {
      float bv = bias ? bias[col] : 0.0f;
#pragma unroll
      for (int tm = 0; tm < 2; ++tm) {
#pragma unroll
        for (int r = 0; r < 8; ++r) {
          int row = mg * 32 + tm * 16 + r + hi * 8;
          if (row < M) {
            float v = acc[tm][tn][r] + bv;
            if (flags & 1) v = fmaxf(v, 0.0f);
            if (Cf) Cf[(size_t)row * ldc + col] = v;
            else    Cb[(size_t)row * ldc + col] = f2bf(v);
          }
        }
      }
    }
  }
}

// Pack f32 weight (K x N row-major) -> bf16 WMMA B-fragment layout, zero-padded.
__global__ void gm_pack_bfrag(const float* __restrict__ W, int K, int Nn,
                              bf16* __restrict__ out, int Ktiles, int Ntiles) {
  long long t = (long long)blockIdx.x * blockDim.x + threadIdx.x;
  long long total = (long long)Ktiles * Ntiles * 512;
  if (t >= total) return;
  int tile = (int)(t >> 9);
  int idx  = (int)(t & 511);
  int n = idx >> 5, k = idx & 31;
  int kt = tile % Ktiles, nt = tile / Ktiles;
  int kg = kt * 32 + k, ng = nt * 16 + n;
  float v = (kg < K && ng < Nn) ? W[(size_t)kg * Nn + ng] : 0.0f;
  out[t] = f2bf(v);
}

// f32 (rows x cols) -> bf16 (rowsPad x colsPad) zero-padded row-major.
__global__ void gm_f32_to_bf16_pad(const float* __restrict__ src, int rows, int cols,
                                   bf16* __restrict__ dst, int rowsPad, int colsPad) {
  long long t = (long long)blockIdx.x * blockDim.x + threadIdx.x;
  long long total = (long long)rowsPad * colsPad;
  if (t >= total) return;
  int r = (int)(t / colsPad), c = (int)(t % colsPad);
  float v = (r < rows && c < cols) ? src[(size_t)r * cols + c] : 0.0f;
  dst[t] = f2bf(v);
}

__global__ void gm_zero_f32(float* p, long long n) {
  long long i = (long long)blockIdx.x * blockDim.x + threadIdx.x;
  if (i < n) p[i] = 0.0f;
}

__global__ void gm_count_deg(const int* __restrict__ dst, float* __restrict__ deg, int E) {
  int e = blockIdx.x * blockDim.x + threadIdx.x;
  if (e < E) atomAddF(&deg[dst[e]], 1.0f);
}

__global__ void gm_make_dinv(const float* __restrict__ deg, float* __restrict__ dinv, int N) {
  int i = blockIdx.x * blockDim.x + threadIdx.x;
  if (i < N) dinv[i] = rsqrtf(1.0f + deg[i]);
}

// agg[dst] += h[src] * dinv[src]*dinv[dst]  (256 feats, float4 per thread)
__global__ void gm_scatter_feat(const int* __restrict__ src, const int* __restrict__ dst,
                                const float* __restrict__ dinv, const float* __restrict__ h,
                                float* __restrict__ agg, int E) {
  long long t = (long long)blockIdx.x * blockDim.x + threadIdx.x;
  int e = (int)(t >> 6);
  if (e >= E) return;
  int f4 = (int)(t & 63);
  int s = src[e], d = dst[e];
  float coef = dinv[s] * dinv[d];
  float4 hv = ((const float4*)(h + (size_t)s * 256))[f4];
  float* ap = agg + (size_t)d * 256 + f4 * 4;
  atomAddF(ap + 0, hv.x * coef);
  atomAddF(ap + 1, hv.y * coef);
  atomAddF(ap + 2, hv.z * coef);
  atomAddF(ap + 3, hv.w * coef);
}

__global__ void gm_scatter_scalar(const int* __restrict__ src, const int* __restrict__ dst,
                                  const float* __restrict__ dinv, const float* __restrict__ hs,
                                  float* __restrict__ aggs, int E) {
  int e = blockIdx.x * blockDim.x + threadIdx.x;
  if (e >= E) return;
  int s = src[e], d = dst[e];
  atomAddF(&aggs[d], hs[s] * dinv[s] * dinv[d]);
}

// x = (acc? x : 0) + relu(agg + h*dinv^2 + bias)
__global__ void gm_gcn_combine(const float* __restrict__ agg, const float* __restrict__ h,
                               const float* __restrict__ dinv, const float* __restrict__ bias,
                               float* __restrict__ x, int N, int acc) {
  int t = blockIdx.x * blockDim.x + threadIdx.x;
  if (t >= N * 256) return;
  int i = t >> 8, f = t & 255;
  float v = agg[t] + h[t] * dinv[i] * dinv[i] + bias[f];
  v = fmaxf(v, 0.0f);
  x[t] = acc ? (x[t] + v) : v;
}

// One wave per row: s1[row] = x.pw,  hs[row] = x.cw
__global__ void gm_rowdots(const float* __restrict__ x, const float* __restrict__ pw,
                           const float* __restrict__ cw, float* __restrict__ s1,
                           float* __restrict__ hs, int N) {
  int w = (int)((blockIdx.x * blockDim.x + threadIdx.x) >> 5);
  int lane = threadIdx.x & 31;
  if (w >= N) return;
  const float* row = x + (size_t)w * 256;
  float a = 0.0f, b = 0.0f;
  for (int i = lane; i < 256; i += 32) {
    float v = row[i];
    a += v * pw[i];
    b += v * cw[i];
  }
  for (int off = 16; off > 0; off >>= 1) {
    a += __shfl_down(a, off, 32);
    b += __shfl_down(b, off, 32);
  }
  if (lane == 0) { s1[w] = a; hs[w] = b; }
}

__global__ void gm_score_combine(const float* __restrict__ s1, const float* __restrict__ aggs,
                                 const float* __restrict__ hs, const float* __restrict__ dinv,
                                 const float* __restrict__ pb, const float* __restrict__ cb,
                                 float* __restrict__ wbuf, int j, int N) {
  int i = blockIdx.x * blockDim.x + threadIdx.x;
  if (i >= N) return;
  float d2 = dinv[i] * dinv[i];
  float s2 = aggs[i] + hs[i] * d2 + cb[0];
  wbuf[i * 3 + j] = 0.9f * (s1[i] + pb[0]) + 0.1f * s2;
}

__global__ void gm_softmax_mix(const float* __restrict__ wbuf, const float* __restrict__ x1,
                               const float* __restrict__ x2, const float* __restrict__ x3,
                               float* __restrict__ nf, float beta, int N) {
  int t = blockIdx.x * blockDim.x + threadIdx.x;
  if (t >= N * 256) return;
  int i = t >> 8;
  float w0 = wbuf[i * 3 + 0], w1 = wbuf[i * 3 + 1], w2 = wbuf[i * 3 + 2];
  float m = fmaxf(fmaxf(w0, w1), w2);
  float e0 = expf(w0 - m), e1 = expf(w1 - m), e2 = expf(w2 - m);
  float inv = 1.0f / (e0 + e1 + e2);
  nf[t] += beta * (e0 * x1[t] + e1 * x2[t] + e2 * x3[t]) * inv;
}

__global__ void gm_emb(const int* __restrict__ sample, const float* __restrict__ nf,
                       float* __restrict__ emb, int B) {
  int t = blockIdx.x * blockDim.x + threadIdx.x;
  if (t >= B * 256) return;
  int b = t >> 8, f = t & 255;
  int i = sample[2 * b], j = sample[2 * b + 1];
  emb[t] = nf[(size_t)i * 256 + f] * nf[(size_t)(NDRC + j) * 256 + f];
}

// ---------------------------------------------------------------------------
// Host orchestration
// ---------------------------------------------------------------------------
static void launch_gemm(hipStream_t s, const bf16* A, int lda, const bf16* Bf,
                        float* Cf, bf16* Cb, int ldc, int M, int Nbound,
                        const float* bias, int Mgroups, int Ngroups, int Ktiles, int flags) {
  int waves = Mgroups * Ngroups;
  int blocks = (waves + 3) / 4;
  gm_wmma_gemm<<<blocks, 128, 0, s>>>(A, lda, Bf, Cf, Cb, ldc, M, Nbound, bias,
                                      Mgroups, Ngroups, Ktiles, flags);
}

static void launch_pack(hipStream_t s, const float* W, int K, int Nn,
                        bf16* out, int Ktiles, int Ntiles) {
  long long total = (long long)Ktiles * Ntiles * 512;
  int blocks = (int)((total + 255) / 256);
  gm_pack_bfrag<<<blocks, 256, 0, s>>>(W, K, Nn, out, Ktiles, Ntiles);
}

static void launch_conv(hipStream_t s, const float* src, int rows, int cols,
                        bf16* dst, int rowsPad, int colsPad) {
  long long total = (long long)rowsPad * colsPad;
  int blocks = (int)((total + 255) / 256);
  gm_f32_to_bf16_pad<<<blocks, 256, 0, s>>>(src, rows, cols, dst, rowsPad, colsPad);
}

static void launch_zero(hipStream_t s, float* p, long long n) {
  int blocks = (int)((n + 255) / 256);
  gm_zero_f32<<<blocks, 256, 0, s>>>(p, n);
}

extern "C" void kernel_launch(void* const* d_in, const int* in_sizes, int n_in,
                              void* d_out, int out_size, void* d_ws, size_t ws_size,
                              hipStream_t stream) {
  const int*   sample = (const int*)d_in[0];
  const float* adj    = (const float*)d_in[1];
  const int*   ei[3]  = {(const int*)d_in[2], (const int*)d_in[3], (const int*)d_in[4]};
  const float* g1w = (const float*)d_in[5];  const float* g1b = (const float*)d_in[6];
  const float* g2w = (const float*)d_in[7];  const float* g2b = (const float*)d_in[8];
  const float* g3w = (const float*)d_in[9];  const float* g3b = (const float*)d_in[10];
  const float* pw  = (const float*)d_in[11]; const float* pb  = (const float*)d_in[12];
  const float* cw  = (const float*)d_in[13]; const float* cb  = (const float*)d_in[14];
  const float* m1w = (const float*)d_in[15]; const float* m1b = (const float*)d_in[16];
  const float* m2w = (const float*)d_in[17]; const float* m2b = (const float*)d_in[18];
  const float* m3w = (const float*)d_in[19]; const float* m3b = (const float*)d_in[20];
  const float* m4w = (const float*)d_in[21]; const float* m4b = (const float*)d_in[22];

  float* out_emb = (float*)d_out;                         // 16384 x 256
  float* out_mlp = (float*)d_out + (size_t)NBC * 256;     // 16384 x 2

  const int N = NN, E = NEC, B = NBC;
  const int Mpad = 2272;                  // ceil(2249/32)*32 (rows, 32-mult for 2-M tiling)
  const int Mgr  = 71;                    // 2272/32
  const int KpadA = 2272, KtlA = 71;      // ceil(2249/32)*32 (inner dim of layer-1)

  char* wsb = (char*)d_ws;
  size_t off = 0;
  auto alloc = [&](size_t bytes) -> char* {
    char* p = wsb + off;
    off += (bytes + 255) & ~(size_t)255;
    return p;
  };

  bf16* adjB   = (bf16*)alloc((size_t)Mpad * KpadA * 2);
  bf16* g1wB0  = (bf16*)alloc((size_t)KtlA * 16 * 512 * 2);
  bf16* g1wB1  = (bf16*)alloc((size_t)KtlA * 16 * 512 * 2);
  bf16* g2wB0  = (bf16*)alloc((size_t)8 * 16 * 512 * 2);
  bf16* g2wB1  = (bf16*)alloc((size_t)8 * 16 * 512 * 2);
  bf16* g3wB0  = (bf16*)alloc((size_t)8 * 16 * 512 * 2);
  bf16* g3wB1  = (bf16*)alloc((size_t)8 * 16 * 512 * 2);
  bf16* m1wB   = (bf16*)alloc((size_t)8 * 64 * 512 * 2);
  bf16* m2wB   = (bf16*)alloc((size_t)32 * 32 * 512 * 2);
  bf16* m3wB   = (bf16*)alloc((size_t)16 * 16 * 512 * 2);
  bf16* m4wB   = (bf16*)alloc((size_t)8 * 4 * 512 * 2);   // N padded to 64 (4 tiles)
  float* h1k[2] = {(float*)alloc((size_t)N * 256 * 4), (float*)alloc((size_t)N * 256 * 4)};
  float* deg    = (float*)alloc((size_t)N * 4);
  float* dinvS[3] = {(float*)alloc((size_t)N * 4), (float*)alloc((size_t)N * 4),
                     (float*)alloc((size_t)N * 4)};
  float* x1  = (float*)alloc((size_t)N * 256 * 4);
  float* x2  = (float*)alloc((size_t)N * 256 * 4);
  float* x3  = (float*)alloc((size_t)N * 256 * 4);
  bf16*  xb  = (bf16*)alloc((size_t)Mpad * 256 * 2);
  float* hbuf = (float*)alloc((size_t)N * 256 * 4);
  float* agg  = (float*)alloc((size_t)N * 256 * 4);
  float* s1v  = (float*)alloc((size_t)N * 4);
  float* hsv  = (float*)alloc((size_t)N * 4);
  float* aggs = (float*)alloc((size_t)N * 4);
  float* wbuf = (float*)alloc((size_t)N * 3 * 4);
  float* nf   = (float*)alloc((size_t)N * 256 * 4);
  bf16*  embB = (bf16*)alloc((size_t)2048 * 256 * 2);
  bf16*  hA   = (bf16*)alloc((size_t)2048 * 1024 * 2);
  bf16*  hB   = (bf16*)alloc((size_t)2048 * 512 * 2);
  bf16*  hC   = (bf16*)alloc((size_t)2048 * 256 * 2);

  // ---- weight / input packing -------------------------------------------
  launch_conv(stream, adj, N, N, adjB, Mpad, KpadA);
  launch_pack(stream, g1w,                   N,   256, g1wB0, KtlA, 16);
  launch_pack(stream, g1w + (size_t)N * 256, N,   256, g1wB1, KtlA, 16);
  launch_pack(stream, g2w,                   256, 256, g2wB0, 8, 16);
  launch_pack(stream, g2w + 256 * 256,       256, 256, g2wB1, 8, 16);
  launch_pack(stream, g3w,                   256, 256, g3wB0, 8, 16);
  launch_pack(stream, g3w + 256 * 256,       256, 256, g3wB1, 8, 16);
  launch_pack(stream, m1w, 256, 1024, m1wB, 8, 64);
  launch_pack(stream, m2w, 1024, 512, m2wB, 32, 32);
  launch_pack(stream, m3w, 512,  256, m3wB, 16, 16);
  launch_pack(stream, m4w, 256,    2, m4wB, 8, 4);

  // ---- degree / dinv per edge set ---------------------------------------
  for (int s = 0; s < 3; ++s) {
    launch_zero(stream, deg, N);
    gm_count_deg<<<(E + 255) / 256, 256, 0, stream>>>(ei[s] + E, deg, E);
    gm_make_dinv<<<(N + 255) / 256, 256, 0, stream>>>(deg, dinvS[s], N);
  }

  // ---- layer-1 GEMMs: h1[k] = adj @ g1w[k]  (shared by all 3 edge sets) --
  launch_gemm(stream, adjB, KpadA, g1wB0, h1k[0], nullptr, 256, N, 256,
              nullptr, Mgr, 4, KtlA, 0);
  launch_gemm(stream, adjB, KpadA, g1wB1, h1k[1], nullptr, 256, N, 256,
              nullptr, Mgr, 4, KtlA, 0);

  const bf16* g2B[2] = {g2wB0, g2wB1};
  const bf16* g3B[2] = {g3wB0, g3wB1};
  const float betas[3] = {0.3f, 0.35f, 0.35f};
  const long long NF = (long long)N * 256;
  const int scatBlocks = (int)(((long long)E * 64 + 255) / 256);

  launch_zero(stream, nf, NF);

  for (int s = 0; s < 3; ++s) {
    const int* esrc = ei[s];
    const int* edst = ei[s] + E;
    const float* dv = dinvS[s];

    // layer 1 -> x1
    for (int k = 0; k < 2; ++k) {
      launch_zero(stream, agg, NF);
      gm_scatter_feat<<<scatBlocks, 256, 0, stream>>>(esrc, edst, dv, h1k[k], agg, E);
      gm_gcn_combine<<<(int)((NF + 255) / 256), 256, 0, stream>>>(
          agg, h1k[k], dv, g1b + k * 256, x1, N, k);
    }
    // layer 2 -> x2
    launch_conv(stream, x1, N, 256, xb, Mpad, 256);
    for (int k = 0; k < 2; ++k) {
      launch_gemm(stream, xb, 256, g2B[k], hbuf, nullptr, 256, N, 256,
                  nullptr, Mgr, 4, 8, 0);
      launch_zero(stream, agg, NF);
      gm_scatter_feat<<<scatBlocks, 256, 0, stream>>>(esrc, edst, dv, hbuf, agg, E);
      gm_gcn_combine<<<(int)((NF + 255) / 256), 256, 0, stream>>>(
          agg, hbuf, dv, g2b + k * 256, x2, N, k);
    }
    // layer 3 -> x3
    launch_conv(stream, x2, N, 256, xb, Mpad, 256);
    for (int k = 0; k < 2; ++k) {
      launch_gemm(stream, xb, 256, g3B[k], hbuf, nullptr, 256, N, 256,
                  nullptr, Mgr, 4, 8, 0);
      launch_zero(stream, agg, NF);
      gm_scatter_feat<<<scatBlocks, 256, 0, stream>>>(esrc, edst, dv, hbuf, agg, E);
      gm_gcn_combine<<<(int)((NF + 255) / 256), 256, 0, stream>>>(
          agg, hbuf, dv, g3b + k * 256, x3, N, k);
    }
    // scores for x1,x2,x3
    const float* xs[3] = {x1, x2, x3};
    for (int j = 0; j < 3; ++j) {
      gm_rowdots<<<(N * 32 + 255) / 256, 256, 0, stream>>>(xs[j], pw, cw, s1v, hsv, N);
      launch_zero(stream, aggs, N);
      gm_scatter_scalar<<<(E + 255) / 256, 256, 0, stream>>>(esrc, edst, dv, hsv, aggs, E);
      gm_score_combine<<<(N + 255) / 256, 256, 0, stream>>>(
          s1v, aggs, hsv, dv, pb, cb, wbuf, j, N);
    }
    // softmax + beta-weighted accumulate into nf
    gm_softmax_mix<<<(int)((NF + 255) / 256), 256, 0, stream>>>(
        wbuf, x1, x2, x3, nf, betas[s], N);
  }

  // ---- embedding gather (first output) ----------------------------------
  gm_emb<<<(B * 256 + 255) / 256, 256, 0, stream>>>(sample, nf, out_emb, B);

  // ---- MLP in 8 chunks of 2048 rows (Mgroups = 64) ----------------------
  for (int c = 0; c < 8; ++c) {
    const float* embSrc = out_emb + (size_t)c * 2048 * 256;
    launch_conv(stream, embSrc, 2048, 256, embB, 2048, 256);
    launch_gemm(stream, embB, 256,  m1wB, nullptr, hA, 1024, 2048, 1024,
                m1b, 64, 16, 8, 1);
    launch_gemm(stream, hA,   1024, m2wB, nullptr, hB, 512,  2048, 512,
                m2b, 64, 8, 32, 1);
    launch_gemm(stream, hB,   512,  m3wB, nullptr, hC, 256,  2048, 256,
                m3b, 64, 4, 16, 1);
    launch_gemm(stream, hC,   256,  m4wB, out_mlp + (size_t)c * 2048 * 2, nullptr,
                2, 2048, 2, m4b, 64, 1, 8, 0);
  }
}
